// GATCrohnModel_26671746908862
// MI455X (gfx1250) — matrix operand order repaired and checked by reading, not compiled, not run on hardware
//
#include <hip/hip_runtime.h>
#include <stdint.h>

#define B_    8
#define N_    2048
#define FIN   64
#define H_    2
#define FOUT  16
#define NEG_SLOPE 0.2f
#define WAVES 4
#define LOG2E 1.44269504088896340736f

typedef __attribute__((ext_vector_type(16))) _Float16 v16h;
typedef __attribute__((ext_vector_type(8)))  _Float16 v8h;
typedef __attribute__((ext_vector_type(4)))  _Float16 v4h;
typedef __attribute__((ext_vector_type(8)))  float    v8f;

__device__ __forceinline__ float lrelu(float x) { return x >= 0.f ? x : NEG_SLOPE * x; }

__device__ __forceinline__ float fexp2(float x) {
#if __has_builtin(__builtin_amdgcn_exp2f)
  return __builtin_amdgcn_exp2f(x);   // v_exp_f32 (2^x native)
#else
  return __builtin_exp2f(x);
#endif
}

__device__ __forceinline__ float wave_max(float v) {
#pragma unroll
  for (int m = 16; m >= 1; m >>= 1) v = fmaxf(v, __shfl_xor(v, m));
  return v;
}

// Kernel 1: h = x@W per node; emit h transposed to f16 as h16T[b][h][f][n]
// (contiguous along n => WMMA B fragment is one contiguous 32B load), plus
// attention logits e_i[b][h][n], e_j[b][h][n].
__global__ void __launch_bounds__(256)
gat_proj_kernel(const float* __restrict__ x,
                const float* __restrict__ W,
                const float* __restrict__ a,
                _Float16* __restrict__ h16T,
                float* __restrict__ eig,
                float* __restrict__ ejg) {
  __shared__ float sW[FIN][H_ * FOUT];   // 8 KB
  __shared__ float sa[H_][2 * FOUT];
  const int tid = threadIdx.x;
  for (int i = tid; i < FIN * H_ * FOUT; i += blockDim.x)
    sW[i / (H_ * FOUT)][i % (H_ * FOUT)] = W[i];
  for (int i = tid; i < H_ * 2 * FOUT; i += blockDim.x)
    sa[i / (2 * FOUT)][i % (2 * FOUT)] = a[i];
  __syncthreads();

  const int t = blockIdx.x * blockDim.x + tid;  // 0 .. B*N-1
  const int b = t / N_, n = t % N_;
  const float4* xr4 = (const float4*)(x + (size_t)t * FIN);

  float acc[H_ * FOUT];
#pragma unroll
  for (int c = 0; c < H_ * FOUT; ++c) acc[c] = 0.f;
#pragma unroll 4
  for (int k4 = 0; k4 < FIN / 4; ++k4) {
    const float4 xv = xr4[k4];
#pragma unroll
    for (int c = 0; c < H_ * FOUT; ++c) {
      acc[c] = fmaf(xv.x, sW[k4 * 4 + 0][c], acc[c]);
      acc[c] = fmaf(xv.y, sW[k4 * 4 + 1][c], acc[c]);
      acc[c] = fmaf(xv.z, sW[k4 * 4 + 2][c], acc[c]);
      acc[c] = fmaf(xv.w, sW[k4 * 4 + 3][c], acc[c]);
    }
  }
#pragma unroll
  for (int hh = 0; hh < H_; ++hh) {
    float ei = 0.f, ej = 0.f;
#pragma unroll
    for (int f = 0; f < FOUT; ++f) {
      const float v = acc[hh * FOUT + f];
      h16T[(size_t)((b * H_ + hh) * FOUT + f) * N_ + n] = (_Float16)v;
      ei = fmaf(v, sa[hh][f], ei);
      ej = fmaf(v, sa[hh][FOUT + f], ej);
    }
    eig[(b * H_ + hh) * N_ + n] = ei;
    ejg[(b * H_ + hh) * N_ + n] = ej;
  }
}

// Kernel 2: per wave, one 16-row n-tile of one batch. All logits kept in the
// log2 domain (e_i, e_j pre-scaled by log2(e); lrelu commutes with positive
// scale), so each exp is a single v_exp_f32.
//   shift = lrelu(eiL + max_j ejL): upper bound on every masked logit =>
//           p = 2^(logit-shift) <= 1, softmax shift-invariance keeps it exact.
//   Fused pass per head: one sweep over adj computes the 16x32 unnormalized
//   p-tile (branchless masking), accumulates row sums in registers, stages the
//   f16 tile in LDS, loads A in contiguous k order (hf*16..+15), and WMMA-accumulates vs h16T.
//   Row sums reduce over 8-lane groups (3x shfl_xor); D rows are scaled by
//   1/sum afterwards. Head H-1 does one extra sweep writing normalized alpha.
__global__ void __launch_bounds__(WAVES * 32)
gat_attn_kernel(const float* __restrict__ adj,
                const _Float16* __restrict__ h16T,
                const float* __restrict__ eig,
                const float* __restrict__ ejg,
                float* __restrict__ out,        // [B][N][FOUT]
                float* __restrict__ alphaout) { // [B][N][N], head H-1
  __shared__ __align__(16) float s_ej[H_][N_];                  // 16 KB (log2-scaled)
  __shared__ __align__(64) _Float16 s_alpha[WAVES][16][32];     // 4 KB
  __shared__ float s_ei[WAVES][16];
  __shared__ float s_sh[WAVES][16];
  __shared__ float s_inv[WAVES][16];
  __shared__ float s_out[WAVES][16][16];                        // 4 KB
  __shared__ float s_red[H_][WAVES];

  const int lane = threadIdx.x & 31;
  const int wave = threadIdx.x >> 5;
  const int tile = blockIdx.x * WAVES + wave;   // 0 .. B*(N/16)-1
  const int b    = tile / (N_ / 16);            // 128 tiles per b, WAVES | 128
  const int n0   = (tile % (N_ / 16)) * 16;

  // block-shared e_j (scaled by log2e); fold in running max while loading
  float m0 = -__builtin_inff(), m1 = -__builtin_inff();
  for (int i = threadIdx.x; i < H_ * N_; i += blockDim.x) {
    const float v = ejg[b * H_ * N_ + i] * LOG2E;
    s_ej[i >> 11][i & (N_ - 1)] = v;
    if ((i >> 11) == 0) m0 = fmaxf(m0, v); else m1 = fmaxf(m1, v);
  }
  m0 = wave_max(m0);
  m1 = wave_max(m1);
  if (lane == 0) { s_red[0][wave] = m0; s_red[1][wave] = m1; }
  __syncthreads();
  float ejmax[H_];
#pragma unroll
  for (int hh = 0; hh < H_; ++hh)
    ejmax[hh] = fmaxf(fmaxf(s_red[hh][0], s_red[hh][1]),
                      fmaxf(s_red[hh][2], s_red[hh][3]));

  const int row = lane & 15;        // A/D fragment lane mapping
  const int hf  = lane >> 4;
  const int rr  = lane >> 3;        // tile compute mapping: 4 rows x 8 lanes
  const int jj  = (lane & 7) << 2;  // each lane covers 4 consecutive j
  float out_acc[8];
#pragma unroll
  for (int v = 0; v < 8; ++v) out_acc[v] = 0.f;

  for (int hh = 0; hh < H_; ++hh) {
    if (lane < 16) {
      const float ei = eig[(b * H_ + hh) * N_ + n0 + lane] * LOG2E;
      s_ei[wave][lane] = ei;
      s_sh[wave][lane] = lrelu(ei + ejmax[hh]);
    }
    asm volatile("s_wait_dscnt 0" ::: "memory");
    float eiL[4], shL[4];
#pragma unroll
    for (int g = 0; g < 4; ++g) {
      eiL[g] = s_ei[wave][g * 4 + rr];
      shL[g] = s_sh[wave][g * 4 + rr];
    }

    // B fragment base: h16T[b][hh][f = lane&15][j], lane covers K = hf*16..+15
    const _Float16* hb =
        h16T + (size_t)((b * H_ + hh) * FOUT + (lane & 15)) * N_ + hf * 16;

    v8f acc = {};
    float psum[4] = {0.f, 0.f, 0.f, 0.f};

    // --- fused: unnormalized p-tile + row sums + WMMA, one adj sweep ---
    for (int c = 0; c < N_ / 32; ++c) {
      const int j0 = c * 32;
      if (c + 1 < N_ / 32) __builtin_prefetch(hb + j0 + 32, 0, 3);
      const float4 ev = *(const float4*)&s_ej[hh][j0 + jj];  // shared by all g
#pragma unroll
      for (int g = 0; g < 4; ++g) {
        const int r = g * 4 + rr;
        const float4 av = *(const float4*)(adj + (size_t)(n0 + r) * N_ + j0 + jj);
        // exps computed unconditionally -> branchless v_cndmask masking
        const float p0 = fexp2(lrelu(eiL[g] + ev.x) - shL[g]);
        const float p1 = fexp2(lrelu(eiL[g] + ev.y) - shL[g]);
        const float p2 = fexp2(lrelu(eiL[g] + ev.z) - shL[g]);
        const float p3 = fexp2(lrelu(eiL[g] + ev.w) - shL[g]);
        const float q0 = (av.x != 0.f) ? p0 : 0.f;
        const float q1 = (av.y != 0.f) ? p1 : 0.f;
        const float q2 = (av.z != 0.f) ? p2 : 0.f;
        const float q3 = (av.w != 0.f) ? p3 : 0.f;
        psum[g] += (q0 + q1) + (q2 + q3);
        v4h hq;
        hq[0] = (_Float16)q0; hq[1] = (_Float16)q1;
        hq[2] = (_Float16)q2; hq[3] = (_Float16)q3;
        *(v4h*)&s_alpha[wave][r][jj] = hq;
      }
      asm volatile("s_wait_dscnt 0" ::: "memory");

      // A fragment: row = lane&15; k in {hf*16..+15} (matches B)
      const _Float16* ar = &s_alpha[wave][row][0];
      const v16h afrag = *(const v16h*)(ar + hf * 16);
      const v16h bfrag = *(const v16h*)(hb + j0);
      acc = __builtin_amdgcn_wmma_f32_16x16x32_f16(false, afrag, false, bfrag,
                                                   (short)0, acc, false, false);
    }

    // row sums: reduce the 8 lanes sharing one rr group, publish 1/sum
#pragma unroll
    for (int g = 0; g < 4; ++g) {
      float s = psum[g];
      s += __shfl_xor(s, 1); s += __shfl_xor(s, 2); s += __shfl_xor(s, 4);
      if ((lane & 7) == 0) s_inv[wave][g * 4 + rr] = 1.f / s;
    }
    asm volatile("s_wait_dscnt 0" ::: "memory");

    // normalize this head's D rows (VGPR v -> row v+hf*8) and accumulate
#pragma unroll
    for (int v = 0; v < 8; ++v) out_acc[v] += acc[v] * s_inv[wave][v + hf * 8];

    // head H-1: one more adj sweep writing normalized alpha (float4)
    if (hh == H_ - 1) {
      for (int r = 0; r < 16; ++r) {
        const float ei  = s_ei[wave][r];
        const float sh  = s_sh[wave][r];
        const float inv = s_inv[wave][r];
        const float4* adjrow = (const float4*)(adj + (size_t)(n0 + r) * N_);
        float4* orow = (float4*)(alphaout + (size_t)(b * N_ + n0 + r) * N_);
        for (int q = lane; q < N_ / 4; q += 32) {
          const float4 av = adjrow[q];
          const float4 ev = *(const float4*)&s_ej[hh][q * 4];
          const float p0 = fexp2(lrelu(ei + ev.x) - sh) * inv;
          const float p1 = fexp2(lrelu(ei + ev.y) - sh) * inv;
          const float p2 = fexp2(lrelu(ei + ev.z) - sh) * inv;
          const float p3 = fexp2(lrelu(ei + ev.w) - sh) * inv;
          float4 o;
          o.x = (av.x != 0.f) ? p0 : 0.f;
          o.y = (av.y != 0.f) ? p1 : 0.f;
          o.z = (av.z != 0.f) ? p2 : 0.f;
          o.w = (av.w != 0.f) ? p3 : 0.f;
          orow[q] = o;
        }
      }
    }
  }

  // mean over heads, stage D (VGPR v -> row v+hf*8, col lane&15), coalesced store
#pragma unroll
  for (int v = 0; v < 8; ++v)
    s_out[wave][v + hf * 8][lane & 15] = out_acc[v] * 0.5f;
  asm volatile("s_wait_dscnt 0" ::: "memory");
  const float* so = &s_out[wave][0][0];
  float* ob = out + (size_t)(b * N_ + n0) * FOUT;
  for (int i = lane; i < 256; i += 32) ob[i] = so[i];
}

extern "C" void kernel_launch(void* const* d_in, const int* in_sizes, int n_in,
                              void* d_out, int out_size, void* d_ws, size_t ws_size,
                              hipStream_t stream) {
  const float* x   = (const float*)d_in[0];
  const float* adj = (const float*)d_in[1];
  const float* W   = (const float*)d_in[2];
  const float* a   = (const float*)d_in[3];

  float* out      = (float*)d_out;                       // [B][N][FOUT]
  float* alphaout = out + (size_t)B_ * N_ * FOUT;        // [B][N][N] head H-1

  char* ws = (char*)d_ws;
  _Float16* h16T = (_Float16*)ws;                                   // 1 MB
  float* eig = (float*)(ws + (size_t)B_ * H_ * FOUT * N_ * sizeof(_Float16));
  float* ejg = eig + (size_t)B_ * H_ * N_;

  gat_proj_kernel<<<(B_ * N_) / 256, 256, 0, stream>>>(x, W, a, h16T, eig, ejg);
  gat_attn_kernel<<<(B_ * (N_ / 16)) / WAVES, WAVES * 32, 0, stream>>>(
      adj, h16T, eig, ejg, out, alphaout);
}
